// SpatialCorrelationSampler_56796647522908
// MI455X (gfx1250) — matrix operand order repaired and checked by reading, not compiled, not run on hardware
//
#include <hip/hip_runtime.h>

// CDNA5 / gfx1250 spatial correlation sampler: banded bf16 WMMA GEMM with
// async-to-LDS row streaming.
// corr[b, dy*21+dx, h, w] = sum_c in1[b,c,h,w] * in2[b,c,h+dy-10, w+dx-10]
//
// Pass 1: transpose both inputs to channel-last bf16 [B,H,W,C] in d_ws.
// Pass 2: block = (16 w-pixels) x (4 h-rows); wave i owns h0+i and all 21 dy.
//         The block streams input2 rows r = h0-10 .. h0+13 through a
//         double-buffered LDS stage using GLOBAL_LOAD_ASYNC_TO_LDS_B128
//         (ASYNCcnt), each row consumed by all 4 waves.  Per row, the three
//         16-col s-tiles run as 3 independent accumulator chains so DS loads
//         overlap the WMMA pipe.

typedef __attribute__((ext_vector_type(16))) __bf16 v16bf;
typedef __attribute__((ext_vector_type(8)))  __bf16 v8bf;
typedef __attribute__((ext_vector_type(8)))  float  v8f;

#define PATCH 21
#define PAD   10
#define NCH   256
#define HH    96
#define WW    96
#define CSTR  (HH * WW)
#define PP    (PATCH * PATCH)

#define NCOLS 48                    // staged s-columns per block: [w0-16, w0+32)
#define CPAD  264                   // padded column stride in bf16 (528B: odd*16B
                                    // -> 16-lane strided b128 reads are bank-free)
#define T_ELEMS   (4 * HH * WW * NCH)
#define T_BYTES   ((size_t)T_ELEMS * 2)
#define WS_NEEDED (2 * T_BYTES)

__device__ __forceinline__ __bf16 f32_to_bf16_rne(float x) {
    unsigned int u = __builtin_bit_cast(unsigned int, x);
    u += 0x7FFFu + ((u >> 16) & 1u);
    return __builtin_bit_cast(__bf16, (unsigned short)(u >> 16));
}

// ---------------------------------------------------------------------------
// CDNA5 async global->LDS copy (VGLOBAL GLOBAL_LOAD_ASYNC_TO_LDS_B128,
// GVS addressing: mem = SGPR_base + VGPR_i32 + ioffset; LDS addr from VGPR).
// Tracked by ASYNCcnt; drained with s_wait_asynccnt before the barrier.
// ---------------------------------------------------------------------------
__device__ __forceinline__ void async_b128_to_lds(unsigned lds_off,
                                                  const void* sbase,
                                                  unsigned byte_off) {
    asm volatile("global_load_async_to_lds_b128 %0, %1, %2"
                 :: "v"(lds_off), "v"(byte_off), "s"(sbase)
                 : "memory");
}
__device__ __forceinline__ void wait_async0() {
    asm volatile("s_wait_asynccnt 0" ::: "memory");
}

// ---------------------------------------------------------------------------
// Pass 1: [B,C,H,W] f32 -> [B,H,W,C] bf16, LDS tile transpose.
// ---------------------------------------------------------------------------
__global__ __launch_bounds__(256)
void transpose_bf16_kernel(const float* __restrict__ in1,
                           const float* __restrict__ in2,
                           __bf16* __restrict__ t1,
                           __bf16* __restrict__ t2) {
    __shared__ float tile[32][33];
    const int w0 = blockIdx.x * 32;
    const int h  = blockIdx.y;
    const int b  = blockIdx.z;
    const int t  = threadIdx.x;

    const float* srcs[2] = { in1, in2 };
    __bf16*      dsts[2] = { t1,  t2  };

    for (int which = 0; which < 2; ++which) {
        const float* src = srcs[which];
        __bf16*      dst = dsts[which];
        for (int cb = 0; cb < 8; ++cb) {
            const int c0 = cb * 32;
            __syncthreads();
            #pragma unroll
            for (int i = 0; i < 4; ++i) {          // lanes sweep w (coalesced)
                const int cc = i * 8 + (t >> 5);
                const int wl = t & 31;
                tile[cc][wl] = src[((b * NCH + c0 + cc) * HH + h) * WW + w0 + wl];
            }
            __syncthreads();
            #pragma unroll
            for (int i = 0; i < 4; ++i) {          // lanes sweep c (coalesced)
                const int idx = i * 256 + t;
                const int cc  = idx & 31;
                const int wl  = idx >> 5;
                dst[(((b * HH + h) * WW) + w0 + wl) * NCH + c0 + cc] =
                    f32_to_bf16_rne(tile[cc][wl]);
            }
        }
    }
}

// One input2 row (48 cols x 256 ch bf16 = 24KB) moved by 128 threads as
// 12 async b128 chunks each.  OOB columns skipped (LDS pre-zeroed there).
__device__ __forceinline__ void row_copy_async(const __bf16* __restrict__ rowp,
                                               unsigned lds_base,
                                               int s0w, int tid) {
    #pragma unroll
    for (int i = 0; i < 12; ++i) {
        const int q = tid + i * 128;       // 1536 chunks: 48 cols * 32 chunks
        const int j = q >> 5, sub = q & 31;
        const int s = s0w + j;
        if (s >= 0 && s < WW)
            async_b128_to_lds(lds_base + (unsigned)((j * CPAD + sub * 8) * 2),
                              rowp,
                              (unsigned)((s * NCH + sub * 8) * 2));
    }
}

__device__ __forceinline__ v16bf ldB(const __bf16* p) {
    const v8bf lo = *(const v8bf*)(p);
    const v8bf hi = *(const v8bf*)(p + 8);
    return __builtin_shufflevector(lo, hi,
               0, 1, 2, 3, 4, 5, 6, 7, 8, 9, 10, 11, 12, 13, 14, 15);
}

// ---------------------------------------------------------------------------
// Pass 2.  Grid (W/16, H/4, B) = (6, 24, 4), block 128 (4 waves).
// ---------------------------------------------------------------------------
__global__ __launch_bounds__(128)
void corr_wmma_lds_kernel(const __bf16* __restrict__ t1,
                          const __bf16* __restrict__ t2,
                          float* __restrict__ out) {
    __shared__ __bf16 sB[2][NCOLS * CPAD];   // 50688 B, double-buffered row
    __shared__ float  sOut[4][16 * 49];      // 12544 B, per-wave extraction

    const int wblk = blockIdx.x;
    const int h0   = blockIdx.y * 4;
    const int b    = blockIdx.z;
    const int w0   = wblk * 16;
    const int s0w  = w0 - 16;

    const int tid  = threadIdx.x;
    const int wave = tid >> 5;
    const int lane = tid & 31;
    const int hf   = lane >> 4;
    const int l16  = lane & 15;
    const int h    = h0 + wave;              // this wave's output row

    // Raw LDS byte offsets (flat LDS aperture keeps the offset in addr[31:0]).
    const unsigned sb0 = (unsigned)(uintptr_t)&sB[0][0];
    const unsigned sb1 = (unsigned)(uintptr_t)&sB[1][0];

    // ---- A = t1[b,h,w0+m,:] (m = l16), resident in registers --------------
    const __bf16* pA = t1 + ((size_t)((b * HH + h) * WW) + w0 + l16) * NCH;
    v16bf A[8];
    #pragma unroll
    for (int kc = 0; kc < 8; ++kc) {
        const int k0 = kc * 32;
        const v8bf lo = *(const v8bf*)(pA + k0 + 16 * hf);
        const v8bf hi = *(const v8bf*)(pA + k0 + 16 * hf + 8);
        A[kc] = __builtin_shufflevector(lo, hi,
                    0, 1, 2, 3, 4, 5, 6, 7, 8, 9, 10, 11, 12, 13, 14, 15);
    }

    // ---- zero both LDS row buffers (covers OOB columns permanently) -------
    {
        const v8bf z = {};
        for (int q = tid; q < 2 * NCOLS * CPAD / 8; q += 128)
            *(v8bf*)(&sB[0][0] + (size_t)q * 8) = z;
    }
    __syncthreads();

    const __bf16* t2b   = t2 + (size_t)b * HH * WW * NCH;
    const int     out_b = b * PP;

    // ---- prologue: stage first row (async) --------------------------------
    {
        const int r0 = h0 - PAD;
        if (r0 >= 0 && r0 < HH)
            row_copy_async(t2b + (size_t)r0 * WW * NCH, sb0, s0w, tid);
    }
    wait_async0();
    __syncthreads();

    // ---- main loop over 24 input2 rows; dy = k - wave (wave-uniform) ------
    for (int k = 0; k < 24; ++k) {
        const int r = h0 - PAD + k;

        // kick off async copy of the next row into the other buffer
        const int  rn     = r + 1;
        const bool nvalid = (k + 1 < 24) && (rn >= 0) && (rn < HH);
        if (nvalid)
            row_copy_async(t2b + (size_t)rn * WW * NCH,
                           ((k + 1) & 1) ? sb1 : sb0, s0w, tid);

        const int dy = k - wave;
        if (dy >= 0 && dy < PATCH) {
            const int obase = ((out_b + dy * PATCH) * HH + h) * WW + w0;
            if (r < 0 || r >= HH) {
                // zero-padding row: coalesced zero stores
                for (int q = lane; q < PATCH * 16; q += 32) {
                    const int dx = q >> 4, m = q & 15;
                    out[obase + dx * CSTR + m] = 0.0f;
                }
            } else {
                const __bf16* bufp = &sB[k & 1][0];
                const __bf16* pB0  = bufp + l16 * CPAD + 16 * hf;
                const __bf16* pB1  = pB0 + 16 * CPAD;
                const __bf16* pB2  = pB0 + 32 * CPAD;

                // 3 independent accumulator chains: 6 DS loads + 3 WMMAs per
                // K-step -> partial dscnt waits, DS latency hides in the pipe.
                v8f acc0 = {}, acc1 = {}, acc2 = {};
                #pragma unroll
                for (int kc = 0; kc < 8; ++kc) {
                    const int o = kc * 32;
                    const v16bf B0 = ldB(pB0 + o);
                    const v16bf B1 = ldB(pB1 + o);
                    const v16bf B2 = ldB(pB2 + o);
                    acc0 = __builtin_amdgcn_wmma_f32_16x16x32_bf16(
                            false, A[kc], false, B0, (short)0, acc0, false, false);
                    acc1 = __builtin_amdgcn_wmma_f32_16x16x32_bf16(
                            false, A[kc], false, B1, (short)0, acc1, false, false);
                    acc2 = __builtin_amdgcn_wmma_f32_16x16x32_bf16(
                            false, A[kc], false, B2, (short)0, acc2, false, false);
                }

                // D layout: lane holds col n, VGPR rr holds row m = rr+8*hf.
                // Stage to per-wave LDS scratch (DS is in-order per wave).
                float* sc = &sOut[wave][0];
                #pragma unroll
                for (int rr = 0; rr < 8; ++rr) {
                    const int mrow = (rr + 8 * hf) * 49;
                    sc[mrow + l16]      = acc0[rr];
                    sc[mrow + 16 + l16] = acc1[rr];
                    sc[mrow + 32 + l16] = acc2[rr];
                }
                // drain band in (dx, m) order -> coalesced 2x64B store runs
                for (int q = lane; q < PATCH * 16; q += 32) {
                    const int dx = q >> 4, m = q & 15;
                    out[obase + dx * CSTR + m] = sc[m * 49 + m + dx + 6];
                }
            }
        }

        wait_async0();          // next row fully landed in LDS
        __syncthreads();
    }
}

// ---------------------------------------------------------------------------
// Fallback (workspace too small): direct fp32->bf16 in the hot loop.
// ---------------------------------------------------------------------------
__global__ __launch_bounds__(128)
void corr_wmma_direct_kernel(const float* __restrict__ in1,
                             const float* __restrict__ in2,
                             float* __restrict__ out) {
    const int wblk = blockIdx.x;
    const int h    = blockIdx.y;
    const int b    = blockIdx.z;
    const int w0   = wblk * 16;
    const int s0   = w0 - 16;
    const int tid  = threadIdx.x;
    const int wave = tid >> 5;
    const int lane = tid & 31;
    const int hf   = lane >> 4;
    const int l16  = lane & 15;

    const float* p1 = in1 + ((b * NCH) * HH + h) * WW + w0 + l16;
    v16bf A[8];
    #pragma unroll
    for (int kc = 0; kc < 8; ++kc) {
        const int k0 = kc * 32;
        #pragma unroll
        for (int e = 0; e < 16; ++e) {
            const int c = k0 + ((e < 8) ? (hf ? 8 + e : e) : (hf ? 16 + e : 8 + e));
            A[kc][e] = f32_to_bf16_rne(p1[c * CSTR]);
        }
    }
    const float* base2 = in2 + (b * NCH) * HH * WW;
    const int    out_b = b * PP;

    for (int dy = wave; dy < PATCH; dy += 4) {
        const int r = h + dy - PAD;
        if (r < 0 || r >= HH) {
            for (int idx = lane; idx < PATCH * 16; idx += 32) {
                const int dx = idx >> 4, m = idx & 15;
                out[((out_b + dy * PATCH + dx) * HH + h) * WW + w0 + m] = 0.0f;
            }
            continue;
        }
        #pragma unroll
        for (int tt = 0; tt < 3; ++tt) {
            const int  s  = s0 + tt * 16 + l16;
            const bool sv = (s >= 0) && (s < WW);
            const int  sc = sv ? s : 0;
            const float* p2 = base2 + r * WW + sc;
            v8f acc = {};
            #pragma unroll
            for (int kc = 0; kc < 8; ++kc) {
                const int k0 = kc * 32;
                v16bf Bv;
                #pragma unroll
                for (int e = 0; e < 16; ++e) {
                    const float x = p2[(k0 + ((e < 8) ? 8 * hf + e : 8 + 8 * hf + e)) * CSTR];
                    Bv[e] = f32_to_bf16_rne(sv ? x : 0.0f);
                }
                acc = __builtin_amdgcn_wmma_f32_16x16x32_bf16(
                        false, A[kc], false, Bv, (short)0, acc, false, false);
            }
            #pragma unroll
            for (int rr = 0; rr < 8; ++rr) {
                const int m  = rr + 8 * hf;
                const int dx = tt * 16 + l16 - m - 6;
                if (dx >= 0 && dx < PATCH) {
                    out[((out_b + dy * PATCH + dx) * HH + h) * WW + w0 + m] = acc[rr];
                }
            }
        }
    }
}

extern "C" void kernel_launch(void* const* d_in, const int* in_sizes, int n_in,
                              void* d_out, int out_size, void* d_ws, size_t ws_size,
                              hipStream_t stream) {
    (void)in_sizes; (void)n_in; (void)out_size;
    const float* in1 = (const float*)d_in[0];
    const float* in2 = (const float*)d_in[1];
    float* out = (float*)d_out;

    if (ws_size >= WS_NEEDED && d_ws != nullptr) {
        __bf16* t1 = (__bf16*)d_ws;
        __bf16* t2 = (__bf16*)((char*)d_ws + T_BYTES);

        hipLaunchKernelGGL(transpose_bf16_kernel, dim3(WW / 32, HH, 4), dim3(256),
                           0, stream, in1, in2, t1, t2);
        hipLaunchKernelGGL(corr_wmma_lds_kernel, dim3(WW / 16, HH / 4, 4),
                           dim3(128), 0, stream, t1, t2, out);
    } else {
        hipLaunchKernelGGL(corr_wmma_direct_kernel, dim3(WW / 16, HH, 4),
                           dim3(128), 0, stream, in1, in2, out);
    }
}
